// Bert_AvgPooling_GCN_77704548319249
// MI455X (gfx1250) — compile-verified, matrix operand order checked by reading, not run on hardware
//
#include <hip/hip_runtime.h>
#include <hip/hip_bf16.h>
#include <cmath>

// ---------------------------------------------------------------------------
// Types for CDNA5 WMMA (wave32): v_wmma_f32_16x16x32_bf16
// ---------------------------------------------------------------------------
typedef __attribute__((ext_vector_type(16))) __bf16 v16bf;
typedef __attribute__((ext_vector_type(8)))  __bf16 v8bf;
typedef __attribute__((ext_vector_type(8)))  float  v8f;

#define D_DIM 768     // hidden dim
#define MWP   128     // padded max words per sentence (real mw = 120)
#define BS    64      // sentences
#define CLAUSE 5

#define WMMA_BF16(A, B, C) \
    __builtin_amdgcn_wmma_f32_16x16x32_bf16(false, (A), false, (B), (short)0, (C), false, false)

// Fragment assembly per ISA 7.12.2 (16-bit A 16x32 layout):
// lane half h: K elements {k0 + h*8 + 0..7} and {k0 + 16 + h*8 + 0..7}
// caller passes p = row_base + h*8 + k0 ; second chunk at p + 16.
__device__ __forceinline__ v16bf make_frag(const __bf16* p) {
    v8bf c1 = *(const v8bf*)(p);
    v8bf c2 = *(const v8bf*)(p + 16);
    v16bf r;
#pragma unroll
    for (int i = 0; i < 8; ++i) { r[i] = c1[i]; r[i + 8] = c2[i]; }
    return r;
}

__device__ __forceinline__ void store_bf8(__bf16* p, v8f a) {
    v8bf o;
#pragma unroll
    for (int v = 0; v < 8; ++v) o[v] = (__bf16)a[v];
    *(v8bf*)p = o;
}

// ---------------------------------------------------------------------------
// Prep kernels
// ---------------------------------------------------------------------------
__global__ void zero_bf16(__bf16* p, int n) {
    int i = blockIdx.x * blockDim.x + threadIdx.x;
    if (i < n) p[i] = (__bf16)0.f;
}

// src[K][N] row-major f32 -> dst[N][K] bf16 (B-operand friendly)
__global__ void transpose_to_bf16(const float* __restrict__ src, __bf16* __restrict__ dst,
                                  int K, int N) {
    int i = blockIdx.x * blockDim.x + threadIdx.x;
    if (i < K * N) {
        int n = i / K, k = i % K;
        dst[(size_t)n * K + k] = (__bf16)src[(size_t)k * N + n];
    }
}

// adj[b][mw][mw] f32 -> adjbf[b][128][128] bf16 (zero padded)
__global__ void adj_pack(const float* __restrict__ adj, __bf16* __restrict__ adjbf, int mw) {
    int i = blockIdx.x * blockDim.x + threadIdx.x;
    if (i >= BS * MWP * MWP) return;
    int b = i / (MWP * MWP);
    int r = (i / MWP) % MWP;
    int c = i % MWP;
    float v = (r < mw && c < mw) ? adj[((size_t)b * mw + r) * mw + c] : 0.f;
    adjbf[i] = (__bf16)v;
}

// inv_denom[b][i] = 1 / (sum_j adj[b][i][j] + 1)
__global__ void inv_denom_k(const float* __restrict__ adj, float* __restrict__ invden, int mw) {
    int i = blockIdx.x * blockDim.x + threadIdx.x;
    if (i >= BS * MWP) return;
    int b = i / MWP, r = i % MWP;
    float s = 0.f;
    if (r < mw) {
        const float* row = adj + ((size_t)b * mw + r) * mw;
        for (int j = 0; j < mw; ++j) s += row[j];
    }
    invden[i] = 1.f / (s + 1.f);
}

// Token gather + per-word segment max -> xbf[(b*128+pos)][d] (bf16, row-major)
__global__ void word_gather_max(const float* __restrict__ seq, const int* __restrict__ tok_idx,
                                const int* __restrict__ word_seg, const int* __restrict__ w2s,
                                const int* __restrict__ wpos, __bf16* __restrict__ xbf, int Nt) {
    int w = blockIdx.x;
    __shared__ int s_lo, s_hi;
    if (threadIdx.x == 0) {
        int lo = 0, hi = Nt;                      // lower_bound(word_seg, w)
        while (lo < hi) { int m = (lo + hi) >> 1; if (word_seg[m] < w) lo = m + 1; else hi = m; }
        s_lo = lo;
        int lo2 = lo, hi2 = Nt;                   // upper_bound(word_seg, w)
        while (lo2 < hi2) { int m = (lo2 + hi2) >> 1; if (word_seg[m] <= w) lo2 = m + 1; else hi2 = m; }
        s_hi = lo2;
    }
    __syncthreads();
    const int t0 = s_lo, t1 = s_hi;
    const int row = w2s[w] * MWP + wpos[w];
    for (int d = threadIdx.x; d < D_DIM; d += blockDim.x) {
        float m = -3.4e38f;
        for (int t = t0; t < t1; ++t)
            m = fmaxf(m, seq[(size_t)tok_idx[t] * D_DIM + d]);
        xbf[(size_t)row * D_DIM + d] = (__bf16)m;
    }
}

// ---------------------------------------------------------------------------
// GEMM1: XW = x @ W.   x: [BS*128, 768] bf16 row-major (A), W^T: [768][768] bf16 (B)
// Output stored TRANSPOSED per sentence: xwt[b][d][j] bf16 (B-operand of GEMM2).
// One wave -> 16 rows x 64 cols (4 WMMA accumulators).
// Ping-pong double-buffered K-loop, unrolled by 2 with peeled tail: both
// fragment buffers are written directly by loads (no register copies).
// ---------------------------------------------------------------------------
__global__ void __launch_bounds__(256) gemm_xw(const __bf16* __restrict__ xbf,
                                               const __bf16* __restrict__ wtbf,
                                               __bf16* __restrict__ xwt) {
    const int lane = threadIdx.x & 31;
    const int wave = threadIdx.x >> 5;
    const int flat = blockIdx.x * 8 + wave;          // 6144 waves total
    const int rowTile = flat / (D_DIM / 64);         // 0..511
    const int colQuad = flat % (D_DIM / 64);         // 0..11
    const int m = lane & 15, h = lane >> 4;

    const __bf16* arow = xbf + (size_t)(rowTile * 16 + m) * D_DIM + h * 8;
    const __bf16* brow = wtbf + (size_t)(colQuad * 64 + m) * D_DIM + h * 8;

    v8f acc[4] = {v8f{}, v8f{}, v8f{}, v8f{}};

    // prologue: P buffer at k = 0
    v16bf aP = make_frag(arow);
    v16bf bP[4];
#pragma unroll
    for (int nt = 0; nt < 4; ++nt) bP[nt] = make_frag(brow + (size_t)nt * 16 * D_DIM);

    // steady state: pairs (k0, k0+32); D_DIM/32 = 24 steps = 12 pairs, last peeled
    for (int k0 = 0; k0 < D_DIM - 64; k0 += 64) {
        __builtin_prefetch(arow + k0 + 64, 0, 1);
        v16bf aQ = make_frag(arow + k0 + 32);
        v16bf bQ[4];
#pragma unroll
        for (int nt = 0; nt < 4; ++nt) bQ[nt] = make_frag(brow + (size_t)nt * 16 * D_DIM + k0 + 32);
#pragma unroll
        for (int nt = 0; nt < 4; ++nt) acc[nt] = WMMA_BF16(aP, bP[nt], acc[nt]);
        aP = make_frag(arow + k0 + 64);
#pragma unroll
        for (int nt = 0; nt < 4; ++nt) bP[nt] = make_frag(brow + (size_t)nt * 16 * D_DIM + k0 + 64);
#pragma unroll
        for (int nt = 0; nt < 4; ++nt) acc[nt] = WMMA_BF16(aQ, bQ[nt], acc[nt]);
    }
    // peeled final pair (k0 = D_DIM-64, D_DIM-32)
    {
        v16bf aQ = make_frag(arow + D_DIM - 32);
        v16bf bQ[4];
#pragma unroll
        for (int nt = 0; nt < 4; ++nt) bQ[nt] = make_frag(brow + (size_t)nt * 16 * D_DIM + D_DIM - 32);
#pragma unroll
        for (int nt = 0; nt < 4; ++nt) acc[nt] = WMMA_BF16(aP, bP[nt], acc[nt]);
#pragma unroll
        for (int nt = 0; nt < 4; ++nt) acc[nt] = WMMA_BF16(aQ, bQ[nt], acc[nt]);
    }

    // D layout: lane -> N = m, rows M = h*8 + v (consecutive) => contiguous b128 store
    const int bIdx = (rowTile * 16) / MWP;
    const int jbase = (rowTile * 16) % MWP + h * 8;
#pragma unroll
    for (int nt = 0; nt < 4; ++nt) {
        int dcol = colQuad * 64 + nt * 16 + m;
        store_bf8(xwt + ((size_t)(bIdx * D_DIM + dcol)) * MWP + jbase, acc[nt]);
    }
}

// ---------------------------------------------------------------------------
// GEMM2: y[b] = relu( (adj[b] @ XW[b]) * inv_denom + bias )
// A = adjbf[b] (128x128 bf16 row-major), B = xwt[b][d][j].  Writes next-layer x
// (bf16 row-major).  Same ping-pong K-loop (4 steps = 1 pair + peeled pair).
// ---------------------------------------------------------------------------
__global__ void __launch_bounds__(256) gemm_adj(const __bf16* __restrict__ adjbf,
                                                const __bf16* __restrict__ xwt,
                                                const float* __restrict__ invden,
                                                const float* __restrict__ bias,
                                                __bf16* __restrict__ xbf) {
    const int lane = threadIdx.x & 31;
    const int wave = threadIdx.x >> 5;
    const int flat = blockIdx.x * 8 + wave;          // 6144 waves total
    const int wavesPerSent = (MWP / 16) * (D_DIM / 64); // 96
    const int b = flat / wavesPerSent;
    const int r = flat % wavesPerSent;
    const int iTile = r / (D_DIM / 64);              // 0..7
    const int colQuad = r % (D_DIM / 64);            // 0..11
    const int m = lane & 15, h = lane >> 4;

    const __bf16* arow = adjbf + ((size_t)(b * MWP + iTile * 16 + m)) * MWP + h * 8;
    const __bf16* bcol = xwt + ((size_t)(b * D_DIM + colQuad * 64 + m)) * MWP + h * 8;

    v8f acc[4] = {v8f{}, v8f{}, v8f{}, v8f{}};

    v16bf aP = make_frag(arow);
    v16bf bP[4];
#pragma unroll
    for (int nt = 0; nt < 4; ++nt) bP[nt] = make_frag(bcol + (size_t)nt * 16 * MWP);

#pragma unroll
    for (int k0 = 0; k0 < MWP - 64; k0 += 64) {
        v16bf aQ = make_frag(arow + k0 + 32);
        v16bf bQ[4];
#pragma unroll
        for (int nt = 0; nt < 4; ++nt) bQ[nt] = make_frag(bcol + (size_t)nt * 16 * MWP + k0 + 32);
#pragma unroll
        for (int nt = 0; nt < 4; ++nt) acc[nt] = WMMA_BF16(aP, bP[nt], acc[nt]);
        aP = make_frag(arow + k0 + 64);
#pragma unroll
        for (int nt = 0; nt < 4; ++nt) bP[nt] = make_frag(bcol + (size_t)nt * 16 * MWP + k0 + 64);
#pragma unroll
        for (int nt = 0; nt < 4; ++nt) acc[nt] = WMMA_BF16(aQ, bQ[nt], acc[nt]);
    }
    {
        v16bf aQ = make_frag(arow + MWP - 32);
        v16bf bQ[4];
#pragma unroll
        for (int nt = 0; nt < 4; ++nt) bQ[nt] = make_frag(bcol + (size_t)nt * 16 * MWP + MWP - 32);
#pragma unroll
        for (int nt = 0; nt < 4; ++nt) acc[nt] = WMMA_BF16(aP, bP[nt], acc[nt]);
#pragma unroll
        for (int nt = 0; nt < 4; ++nt) acc[nt] = WMMA_BF16(aQ, bQ[nt], acc[nt]);
    }

    float idv[8];
#pragma unroll
    for (int v = 0; v < 8; ++v) idv[v] = invden[b * MWP + iTile * 16 + h * 8 + v];
#pragma unroll
    for (int nt = 0; nt < 4; ++nt) {
        int d = colQuad * 64 + nt * 16 + m;
        float bi = bias[d];
#pragma unroll
        for (int v = 0; v < 8; ++v) {
            int i = iTile * 16 + h * 8 + v;
            float val = fmaxf(acc[nt][v] * idv[v] + bi, 0.f);
            xbf[((size_t)(b * MWP + i)) * D_DIM + d] = (__bf16)val;
        }
    }
}

// ---------------------------------------------------------------------------
// Clause segment-max (5 words per clause); zero-pads rows up to Mpad for WMMA FC.
// ---------------------------------------------------------------------------
__global__ void clause_max(const __bf16* __restrict__ xbf, const int* __restrict__ w2s,
                           const int* __restrict__ wpos, __bf16* __restrict__ cbf,
                           int Wt, int nC) {
    int c = blockIdx.x;
    if (c >= nC) {
        for (int d = threadIdx.x; d < D_DIM; d += blockDim.x)
            cbf[(size_t)c * D_DIM + d] = (__bf16)0.f;
        return;
    }
    int w0 = c * CLAUSE;
    int n = min(w0 + CLAUSE, Wt) - w0;
    int rows[CLAUSE];
    for (int k = 0; k < n; ++k) rows[k] = w2s[w0 + k] * MWP + wpos[w0 + k];
    for (int d = threadIdx.x; d < D_DIM; d += blockDim.x) {
        float mval = -3.4e38f;
        for (int k = 0; k < n; ++k)
            mval = fmaxf(mval, (float)xbf[(size_t)rows[k] * D_DIM + d]);
        cbf[(size_t)c * D_DIM + d] = (__bf16)mval;
    }
}

// ---------------------------------------------------------------------------
// FC: logits = clause_out @ Wfc + bfc.  N = 16 == one WMMA tile. f32 out.
// Ping-pong K-loop (24 steps = 11 pairs + peeled pair).
// ---------------------------------------------------------------------------
__global__ void __launch_bounds__(256) gemm_fc(const __bf16* __restrict__ cbf,
                                               const __bf16* __restrict__ wfct,
                                               const float* __restrict__ bfc,
                                               float* __restrict__ out,
                                               int nC, int rowTiles) {
    const int lane = threadIdx.x & 31;
    const int wave = threadIdx.x >> 5;
    const int tile = blockIdx.x * 8 + wave;
    if (tile >= rowTiles) return;                    // wave-uniform
    const int m = lane & 15, h = lane >> 4;

    const __bf16* arow = cbf + (size_t)(tile * 16 + m) * D_DIM + h * 8;
    const __bf16* bcol = wfct + (size_t)m * D_DIM + h * 8;

    v8f acc = {};
    v16bf aP = make_frag(arow);
    v16bf bP = make_frag(bcol);
    for (int k0 = 0; k0 < D_DIM - 64; k0 += 64) {
        v16bf aQ = make_frag(arow + k0 + 32);
        v16bf bQ = make_frag(bcol + k0 + 32);
        acc = WMMA_BF16(aP, bP, acc);
        aP = make_frag(arow + k0 + 64);
        bP = make_frag(bcol + k0 + 64);
        acc = WMMA_BF16(aQ, bQ, acc);
    }
    {
        v16bf aQ = make_frag(arow + D_DIM - 32);
        v16bf bQ = make_frag(bcol + D_DIM - 32);
        acc = WMMA_BF16(aP, bP, acc);
        acc = WMMA_BF16(aQ, bQ, acc);
    }
    float bi = bfc[m];
#pragma unroll
    for (int v = 0; v < 8; ++v) {
        int row = tile * 16 + h * 8 + v;
        if (row < nC) out[(size_t)row * 16 + m] = acc[v] + bi;
    }
}

// ---------------------------------------------------------------------------
// Host launch
// ---------------------------------------------------------------------------
extern "C" void kernel_launch(void* const* d_in, const int* in_sizes, int n_in,
                              void* d_out, int out_size, void* d_ws, size_t ws_size,
                              hipStream_t stream) {
    const float* seq  = (const float*)d_in[0];
    const float* adj  = (const float*)d_in[1];
    const float* W[3]  = {(const float*)d_in[2], (const float*)d_in[4], (const float*)d_in[6]};
    const float* bb[3] = {(const float*)d_in[3], (const float*)d_in[5], (const float*)d_in[7]};
    const float* Wfc = (const float*)d_in[8];
    const float* bfc = (const float*)d_in[9];
    const int* tok_idx  = (const int*)d_in[10];
    const int* word_seg = (const int*)d_in[11];
    const int* w2s      = (const int*)d_in[12];
    const int* wpos     = (const int*)d_in[13];

    const int Nt = in_sizes[10];
    const int Wt = in_sizes[12];
    const int mw = (int)(std::sqrt((double)in_sizes[1] / BS) + 0.5);
    const int nC = out_size / 16;
    const int rowTiles = (nC + 15) / 16;
    const int Mpad = rowTiles * 16;

    // workspace carve-out (256B aligned)
    char* p = (char*)d_ws;
    auto carve = [&](size_t bytes) -> char* {
        char* r = p; p += ((bytes + 255) / 256) * 256; return r;
    };
    __bf16* xbf   = (__bf16*)carve((size_t)BS * MWP * D_DIM * 2);   // x, row-major
    __bf16* xwt   = (__bf16*)carve((size_t)BS * D_DIM * MWP * 2);   // (x@W)^T per sentence
    __bf16* adjbf = (__bf16*)carve((size_t)BS * MWP * MWP * 2);
    float*  invd  = (float*) carve((size_t)BS * MWP * 4);
    __bf16* wtbf  = (__bf16*)carve((size_t)3 * D_DIM * D_DIM * 2);  // W^T bf16 x3
    __bf16* wfct  = (__bf16*)carve((size_t)16 * D_DIM * 2);         // Wfc^T bf16
    __bf16* cbf   = (__bf16*)carve((size_t)Mpad * D_DIM * 2);
    (void)ws_size; (void)n_in;

    // weight conversion (transposed bf16)
    for (int l = 0; l < 3; ++l)
        transpose_to_bf16<<<(D_DIM * D_DIM + 255) / 256, 256, 0, stream>>>(
            W[l], wtbf + (size_t)l * D_DIM * D_DIM, D_DIM, D_DIM);
    transpose_to_bf16<<<(D_DIM * 16 + 255) / 256, 256, 0, stream>>>(Wfc, wfct, D_DIM, 16);

    // padded x init + adjacency prep
    zero_bf16<<<(BS * MWP * D_DIM + 255) / 256, 256, 0, stream>>>(xbf, BS * MWP * D_DIM);
    adj_pack<<<(BS * MWP * MWP + 255) / 256, 256, 0, stream>>>(adj, adjbf, mw);
    inv_denom_k<<<(BS * MWP + 255) / 256, 256, 0, stream>>>(adj, invd, mw);

    // token gather + word max
    word_gather_max<<<Wt, 256, 0, stream>>>(seq, tok_idx, word_seg, w2s, wpos, xbf, Nt);

    // 3 GCN layers: GEMM1 (6144 waves) + GEMM2 (6144 waves)
    const int gemmBlocks = (BS * MWP / 16) * (D_DIM / 64) / 8;  // 768
    for (int l = 0; l < 3; ++l) {
        gemm_xw<<<gemmBlocks, 256, 0, stream>>>(xbf, wtbf + (size_t)l * D_DIM * D_DIM, xwt);
        gemm_adj<<<gemmBlocks, 256, 0, stream>>>(adjbf, xwt, invd, bb[l], xbf);
    }

    // clause max + FC
    clause_max<<<Mpad, 256, 0, stream>>>(xbf, w2s, wpos, cbf, Wt, nC);
    gemm_fc<<<(rowTiles + 7) / 8, 256, 0, stream>>>(cbf, wfct, bfc, (float*)d_out, nC, rowTiles);
}
